// HeteroGNN_31696858644809
// MI455X (gfx1250) — matrix operand order repaired and checked by reading, not compiled, not run on hardware
//
#include <hip/hip_runtime.h>
#include <hip/hip_bf16.h>

typedef __attribute__((ext_vector_type(16))) _Float16 v16h;
typedef __attribute__((ext_vector_type(8)))  _Float16 v8h;
typedef __attribute__((ext_vector_type(4)))  _Float16 v4h;
typedef __attribute__((ext_vector_type(8)))  float    v8f;

#define NNODE 100000
#define NEDGE 1600000
#define HDIM  64

// ---------------------------------------------------------------------------
// WMMA GEMM: C[N, NT*16] = A[N, K] @ W[K, NT*16]   (f32 in/out, f16 compute)
// Block = 256 threads = 8 waves; wave w computes rows [blk*128 + w*16, +16).
// A tile and W^T are staged in LDS as f16 (coalesced global loads, one
// conversion per element per block); fragments are built with ds_load_b128.
// Padded stride 72 f16 (144B) keeps 16B alignment and rotates LDS banks.
// ---------------------------------------------------------------------------
template <int NT>
__global__ __launch_bounds__(256) void k_gemm_wmma(
    const float* __restrict__ A, const float* __restrict__ W,
    float* __restrict__ C, int N, int K) {
  constexpr int NCOL = NT * 16;
  constexpr int LDA  = 72;                 // f16 stride: max K (64) + 8 pad
  __shared__ _Float16 Ash[128 * LDA];      // 18 KB
  __shared__ _Float16 WshT[NCOL * LDA];    // up to 18 KB (NT=8)

  const int tid  = threadIdx.x;
  const int wave = tid >> 5;
  const int lane = tid & 31;
  const int rowBase = blockIdx.x * 128 + wave * 16;
  const int kh  = lane >> 4;               // half-wave select
  const int l15 = lane & 15;

  // ---- stage A tile [128 x K] -> f16 LDS (rows are contiguous in A) ----
  {
    const size_t aBase = (size_t)blockIdx.x * 128 * K;
    const int nq = (128 * K) / 4;
    for (int i = tid; i < nq; i += 256) {
      const int row = (i * 4) / K;
      const int kk  = (i * 4) % K;
      float4 f = make_float4(0.f, 0.f, 0.f, 0.f);
      if (blockIdx.x * 128 + row < N) f = ((const float4*)(A + aBase))[i];
      v4h h;
      h[0] = (_Float16)f.x; h[1] = (_Float16)f.y;
      h[2] = (_Float16)f.z; h[3] = (_Float16)f.w;
      *(v4h*)(Ash + row * LDA + kk) = h;
    }
  }
  // ---- stage W^T [NCOL x K] -> f16 LDS (k-contiguous per output column) ----
  for (int i = tid; i < K * NCOL; i += 256) {
    const int k = i / NCOL, c = i % NCOL;
    WshT[c * LDA + k] = (_Float16)W[i];
  }
  __syncthreads();

  if (rowBase + 16 > N) return;            // wave-uniform: EXEC all-ones

  v8f acc[NT];
  v8f zero = {};
#pragma unroll
  for (int i = 0; i < NT; ++i) acc[i] = zero;

  for (int k0 = 0; k0 < K; k0 += 32) {
    // A fragment: lane<16 -> row=l15, K=k0+{0..7,16..23}; lane>=16 -> +8
    const _Float16* ap = Ash + (wave * 16 + l15) * LDA + k0 + kh * 8;
    const v8h a0 = *(const v8h*)ap;
    const v8h a1 = *(const v8h*)(ap + 16);
    v16h afrag;
#pragma unroll
    for (int i = 0; i < 8; ++i) { afrag[i] = a0[i]; afrag[8 + i] = a1[i]; }

#pragma unroll
    for (int nt = 0; nt < NT; ++nt) {
      // B fragment: lane<16 -> col=l15, K=k0+{0..7,16..23}; lane>=16 -> +8
      const _Float16* bp = WshT + (nt * 16 + l15) * LDA + k0 + kh * 8;
      const v8h b0 = *(const v8h*)bp;
      const v8h b1 = *(const v8h*)(bp + 16);
      v16h bfrag;
#pragma unroll
      for (int i = 0; i < 8; ++i) { bfrag[i] = b0[i]; bfrag[8 + i] = b1[i]; }
      acc[nt] = __builtin_amdgcn_wmma_f32_16x16x32_f16(
          false, afrag, false, bfrag, (short)0, acc[nt], false, false);
    }
  }

  // C layout: VGPR r -> row rowBase + r + (lane>=16 ? 8 : 0), col = l15
  const int rbase = rowBase + kh * 8;
#pragma unroll
  for (int nt = 0; nt < NT; ++nt) {
    const int col = nt * 16 + l15;
#pragma unroll
    for (int r = 0; r < 8; ++r)
      C[(size_t)(rbase + r) * NCOL + col] = acc[nt][r];
  }
}

// out[n] = dot(M[n, 0:K], v[0:K])
__global__ void k_rowdot(const float* __restrict__ M, const float* __restrict__ v,
                         float* __restrict__ out, int N, int K) {
  int n = blockIdx.x * blockDim.x + threadIdx.x;
  if (n >= N) return;
  const float4* r  = (const float4*)(M + (size_t)n * K);
  const float4* vv = (const float4*)v;
  float s = 0.f;
  for (int i = 0; i < K / 4; ++i) {
    float4 a = r[i], b = vv[i];
    s += a.x * b.x + a.y * b.y + a.z * b.z + a.w * b.w;
  }
  out[n] = s;
}

// wvec[t] = sum_j W_edge[t,j] * att_edge[j]   (16x64 @ 64)
__global__ void k_colvec16(const float* __restrict__ Wedge,
                           const float* __restrict__ attE,
                           float* __restrict__ wvec) {
  int t = threadIdx.x;
  if (t >= 16) return;
  float s = 0.f;
  for (int j = 0; j < 64; ++j) s += Wedge[t * 64 + j] * attE[j];
  wvec[t] = s;
}

__device__ __forceinline__ unsigned enc_f32(float f) {
  unsigned b = __float_as_uint(f);
  return (b & 0x80000000u) ? ~b : (b | 0x80000000u);
}
__device__ __forceinline__ float dec_f32(unsigned k) {
  unsigned b = (k & 0x80000000u) ? (k & 0x7FFFFFFFu) : ~k;
  return __uint_as_float(b);
}

// Pass 1 (fused edge-attn dot): logit = leaky_relu(a_s[src]+a_d[dst]+ea.wvec);
// segment max via order-preserving uint atomicMax (memset-0 == below -inf).
__global__ void k_edge_logit(const int* __restrict__ ei,
                             const float* __restrict__ a_s,
                             const float* __restrict__ a_d,
                             const float* __restrict__ ea,
                             const float* __restrict__ wvec,
                             float* __restrict__ elog,
                             unsigned* __restrict__ m, int E) {
  int e = blockIdx.x * blockDim.x + threadIdx.x;
  if (e >= E) return;
  int src = ei[e], dst = ei[E + e];
  const float4* r  = (const float4*)(ea + (size_t)e * 16);
  const float4* vv = (const float4*)wvec;
  float s = a_s[src] + a_d[dst];
#pragma unroll
  for (int i = 0; i < 4; ++i) {
    float4 a = r[i], b = vv[i];
    s += a.x * b.x + a.y * b.y + a.z * b.z + a.w * b.w;
  }
  float l = (s > 0.f) ? s : 0.2f * s;
  elog[e] = l;
  atomicMax(m + dst, enc_f32(l));
}

// Pass 2: e = exp(logit - m[dst]); den[dst] += e
__global__ void k_edge_exp(const int* __restrict__ ei,
                           float* __restrict__ elog,
                           const unsigned* __restrict__ m,
                           float* __restrict__ den, int E) {
  int e = blockIdx.x * blockDim.x + threadIdx.x;
  if (e >= E) return;
  int dst = ei[E + e];
  float ex = __expf(elog[e] - dec_f32(m[dst]));
  elog[e] = ex;
  atomicAdd(den + dst, ex);
}

// Pass 3: agg[dst] += hs[src] * alpha ; 16 threads per edge, 4 channels each
__global__ void k_edge_scatter(const int* __restrict__ ei,
                               const float* __restrict__ elog,
                               const float* __restrict__ den,
                               const float* __restrict__ hs,
                               float* __restrict__ agg, int E) {
  long long tid = (long long)blockIdx.x * blockDim.x + threadIdx.x;
  int e = (int)(tid >> 4);
  if (e >= E) return;
  int sub = (int)(tid & 15);
  int src = ei[e], dst = ei[E + e];
  float alpha = elog[e] / (den[dst] + 1e-16f);
  float4 h = ((const float4*)(hs + (size_t)src * HDIM))[sub];
  float* o = agg + (size_t)dst * HDIM + sub * 4;
  atomicAdd(o + 0, h.x * alpha);
  atomicAdd(o + 1, h.y * alpha);
  atomicAdd(o + 2, h.z * alpha);
  atomicAdd(o + 3, h.w * alpha);
}

// out = bn(relu(in + bias))  (eval-mode BatchNorm1d), H must be a power of 2
__global__ void k_bias_relu_bn(const float* __restrict__ in,
                               const float* __restrict__ bias,
                               const float* __restrict__ gamma,
                               const float* __restrict__ beta,
                               const float* __restrict__ mean,
                               const float* __restrict__ var,
                               float* __restrict__ out, int N, int H) {
  long long i = (long long)blockIdx.x * blockDim.x + threadIdx.x;
  if (i >= (long long)N * H) return;
  int c = (int)(i & (H - 1));
  float v = in[i] + bias[c];
  v = fmaxf(v, 0.f);
  out[i] = (v - mean[c]) * rsqrtf(var[c] + 1e-5f) * gamma[c] + beta[c];
}

// out[n] = sigmoid(dot(hid[n,0:128], W2) + b2)
__global__ void k_lin2(const float* __restrict__ hid,
                       const float* __restrict__ W2,
                       const float* __restrict__ b2,
                       float* __restrict__ out, int N) {
  int n = blockIdx.x * blockDim.x + threadIdx.x;
  if (n >= N) return;
  const float4* r = (const float4*)(hid + (size_t)n * 128);
  const float4* w = (const float4*)W2;
  float s = 0.f;
  for (int i = 0; i < 32; ++i) {
    float4 a = r[i], b = w[i];
    s += a.x * b.x + a.y * b.y + a.z * b.z + a.w * b.w;
  }
  float y = s + b2[0];
  out[n] = 1.f / (1.f + __expf(-y));
}

extern "C" void kernel_launch(void* const* d_in, const int* in_sizes, int n_in,
                              void* d_out, int out_size, void* d_ws, size_t ws_size,
                              hipStream_t stream) {
  (void)in_sizes; (void)out_size; (void)ws_size;
  const float* x_claim = (const float*)d_in[0];
  const float* x_party = (const float*)d_in[1];
  const int*   ei_cp   = (const int*)d_in[2];
  const float* ea_cp   = (const float*)d_in[3];
  const int*   ei_pc   = (const int*)d_in[4];
  const float* ea_pc   = (const float*)d_in[5];
  const float* P[64];
  for (int i = 0; i < n_in && i < 64; ++i) P[i] = (const float*)d_in[i];
  // params flatten (insertion order): per layer (22 entries starting at 6):
  //  cp{W_src,W_dst,W_edge,att_src,att_dst,att_edge,bias}=+0..6,
  //  pc{...}=+7..13, bn_claim{g,b,m,v}=+14..17, bn_party=+18..21
  // then 50:lin1_W 51:lin1_b 52..55:lin_bn 56:lin2_W 57:lin2_b

  // ---- workspace layout (floats) ----
  float* w = (float*)d_ws;
  float* xc   = w;  w += (size_t)NNODE * HDIM;
  float* xp   = w;  w += (size_t)NNODE * HDIM;
  float* hs   = w;  w += (size_t)NNODE * HDIM;   // hid[N,128] reuses hs+hd
  float* hd   = w;  w += (size_t)NNODE * HDIM;
  float* hid  = hs;
  float* aggC = w;  w += (size_t)NNODE * HDIM;
  float* aggP = w;  w += (size_t)NNODE * HDIM;
  float* a_s  = w;  w += NNODE;
  float* a_d  = w;  w += NNODE;
  unsigned* mmax = (unsigned*)w; w += NNODE;
  float* den  = w;  w += NNODE;
  float* elog = w;  w += NEDGE;
  float* wvec = w;  w += 16;

  const int gemmGrid = (NNODE + 127) / 128;
  const int nodeGrid = (NNODE + 255) / 256;
  const int edgeGrid = (NEDGE + 255) / 256;
  const int scatGrid = (int)(((long long)NEDGE * 16 + 255) / 256);
  const int elemGrid = (NNODE * HDIM) / 256;

  auto conv = [&](const float* xsrc, int Ksrc, const float* xdst, int Kdst,
                  const int* ei, const float* ea, const float* const* pp,
                  float* aggOut) {
    // pp: W_src, W_dst, W_edge, att_src, att_dst, att_edge
    k_gemm_wmma<4><<<gemmGrid, 256, 0, stream>>>(xsrc, pp[0], hs, NNODE, Ksrc);
    k_rowdot<<<nodeGrid, 256, 0, stream>>>(hs, pp[3], a_s, NNODE, HDIM);
    k_gemm_wmma<4><<<gemmGrid, 256, 0, stream>>>(xdst, pp[1], hd, NNODE, Kdst);
    k_rowdot<<<nodeGrid, 256, 0, stream>>>(hd, pp[4], a_d, NNODE, HDIM);
    k_colvec16<<<1, 64, 0, stream>>>(pp[2], pp[5], wvec);
    hipMemsetAsync(mmax, 0, NNODE * sizeof(unsigned), stream);
    hipMemsetAsync(den, 0, NNODE * sizeof(float), stream);
    hipMemsetAsync(aggOut, 0, (size_t)NNODE * HDIM * sizeof(float), stream);
    k_edge_logit<<<edgeGrid, 256, 0, stream>>>(ei, a_s, a_d, ea, wvec, elog,
                                               mmax, NEDGE);
    k_edge_exp<<<edgeGrid, 256, 0, stream>>>(ei, elog, mmax, den, NEDGE);
    k_edge_scatter<<<scatGrid, 256, 0, stream>>>(ei, elog, den, hs, aggOut,
                                                 NEDGE);
  };

  const float* curC = x_claim; int KC = 64;
  const float* curP = x_party; int KP = 32;
  for (int l = 0; l < 2; ++l) {
    const int b = 6 + l * 22;
    const float* cp[6] = {P[b+0], P[b+1], P[b+2], P[b+3], P[b+4], P[b+5]};
    const float* pc[6] = {P[b+7], P[b+8], P[b+9], P[b+10], P[b+11], P[b+12]};
    // claim -> party (uses old xc, xp) into aggP
    conv(curC, KC, curP, KP, ei_cp, ea_cp, cp, aggP);
    // party -> claim into aggC
    conv(curP, KP, curC, KC, ei_pc, ea_pc, pc, aggC);
    // xc = bn_claim(relu(aggC + bias_pc)) ; xp = bn_party(relu(aggP + bias_cp))
    k_bias_relu_bn<<<elemGrid, 256, 0, stream>>>(
        aggC, P[b+13], P[b+14], P[b+15], P[b+16], P[b+17], xc, NNODE, HDIM);
    k_bias_relu_bn<<<elemGrid, 256, 0, stream>>>(
        aggP, P[b+6],  P[b+18], P[b+19], P[b+20], P[b+21], xp, NNODE, HDIM);
    curC = xc; KC = 64; curP = xp; KP = 64;
  }

  // readout: hid = xc @ lin1_W [64x128]; bn(relu(+b)); sigmoid(hid @ lin2_W + b)
  k_gemm_wmma<8><<<gemmGrid, 256, 0, stream>>>(xc, P[50], hid, NNODE, 64);
  k_bias_relu_bn<<<elemGrid * 2, 256, 0, stream>>>(
      hid, P[51], P[52], P[53], P[54], P[55], hid, NNODE, 128);
  k_lin2<<<nodeGrid, 256, 0, stream>>>(hid, P[56], P[57], (float*)d_out, NNODE);
}
